// Encoder_45329084842195
// MI455X (gfx1250) — matrix operand order repaired and checked by reading, not compiled, not run on hardware
//
#include <hip/hip_runtime.h>
#include <hip/hip_bf16.h>

typedef __attribute__((ext_vector_type(16))) _Float16 v16h;
typedef __attribute__((ext_vector_type(8)))  _Float16 v8h;
typedef __attribute__((ext_vector_type(4)))  _Float16 v4h;
typedef __attribute__((ext_vector_type(8)))  float    v8f;
typedef __attribute__((ext_vector_type(4)))  float    v4f;

#define BSZ   64
#define SSQ   1024
#define IIN   512
#define HHD   512
#define HCOLS 16
#define CGRP  (HHD / HCOLS)     // 32 column-groups per direction
#define NWG   (2 * CGRP)        // 64 persistent workgroups
#define KTOT  (IIN + HHD)       // 1024
#define NROWS (4 * HCOLS)       // 64 gate rows per WG slice

// ---- workspace layout (bytes) ----
#define WS_XH     ((size_t)0)                                   // [SS][BS][I] f16   : 67,108,864 B
#define WS_WP     ((size_t)67108864)                            // [NWG][64][1024] f16: 8,388,608 B
#define WS_BP     (WS_WP + (size_t)8388608)                     // [NWG][64] f32     : 16,384 B
#define WS_HST    (WS_BP + (size_t)16384)                       // [2][2][BS][H] f16 : 524,288 B
#define WS_SYNC   (WS_HST + (size_t)524288)                     // u32 counter

union V16U { v16h v; struct { v8h lo; v8h hi; } p; };

__device__ __forceinline__ float fsigmoid(float x) {
    return 1.0f / (1.0f + __expf(-x));
}
__device__ __forceinline__ float ftanh(float x) {
    return 1.0f - 2.0f / (__expf(2.0f * x) + 1.0f);
}

// ---------------- prep kernels ----------------

// zero h-state double buffer (131072 dwords) + sync counter
__global__ void k_init(unsigned* hst_u32, unsigned* sync) {
    int i = blockIdx.x * 256 + threadIdx.x;
    if (i < 131072) hst_u32[i] = 0u;
    if (i == 0) *sync = 0u;
}

// X [BS][SS][I] f32 -> Xh [SS][BS][I] f16
__global__ void k_convert_x(const float* __restrict__ X, _Float16* __restrict__ Xh) {
    int idx = blockIdx.x * 256 + threadIdx.x;          // 8,388,608 threads
    int e   = idx * 4;
    int i   = e & 511;
    int tb  = e >> 9;
    int b   = tb & 63;
    int t   = tb >> 6;
    v4f src = *(const v4f*)(X + (((size_t)b * SSQ + t) << 9) + i);
    v4h dst;
    dst.x = (_Float16)src.x; dst.y = (_Float16)src.y;
    dst.z = (_Float16)src.z; dst.w = (_Float16)src.w;
    *(v4h*)(Xh + (size_t)e) = dst;
}

// pack weight slices: Wp[wg][r][k], r = gate*16 + j (h-col j of this group), k: [W_ih | W_hh]
__global__ void k_pack_w(const float* __restrict__ Wih_f, const float* __restrict__ Whh_f,
                         const float* __restrict__ Wih_b, const float* __restrict__ Whh_b,
                         _Float16* __restrict__ Wp) {
    int idx = blockIdx.x * 256 + threadIdx.x;          // 1,048,576 threads
    int e   = idx * 4;
    int k   = e & 1023;
    int r   = (e >> 10) & 63;
    int sl  = e >> 16;                                 // 0..63 (dir*32+cg)
    int dir = sl >> 5;
    int cg  = sl & 31;
    int g   = r >> 4;
    int jj  = r & 15;
    int grow = g * HHD + cg * HCOLS + jj;              // row in [4H, K] weight
    const float* Wih = dir ? Wih_b : Wih_f;
    const float* Whh = dir ? Whh_b : Whh_f;
    const float* src = (k < IIN) ? (Wih + (size_t)grow * IIN + k)
                                 : (Whh + (size_t)grow * HHD + (k - IIN));
    v4f s = *(const v4f*)src;
    v4h d;
    d.x = (_Float16)s.x; d.y = (_Float16)s.y; d.z = (_Float16)s.z; d.w = (_Float16)s.w;
    *(v4h*)(Wp + (size_t)e) = d;
}

__global__ void k_pack_bias(const float* __restrict__ bih_f, const float* __restrict__ bhh_f,
                            const float* __restrict__ bih_b, const float* __restrict__ bhh_b,
                            float* __restrict__ Bp) {
    int idx = blockIdx.x * 256 + threadIdx.x;          // 4096 threads
    if (idx >= NWG * NROWS) return;
    int r   = idx & 63;
    int sl  = idx >> 6;
    int dir = sl >> 5;
    int cg  = sl & 31;
    int g   = r >> 4;
    int jj  = r & 15;
    int grow = g * HHD + cg * HCOLS + jj;
    Bp[idx] = dir ? (bih_b[grow] + bhh_b[grow]) : (bih_f[grow] + bhh_f[grow]);
}

// ---------------- persistent bidirectional LSTM ----------------

__global__ void lstm_persistent(const _Float16* __restrict__ Xh,
                                const _Float16* __restrict__ Wp,
                                const float*    __restrict__ Bp,
                                _Float16*       __restrict__ Hst,
                                float*          __restrict__ Outs,
                                float*          __restrict__ Hs,
                                float*          __restrict__ Cs,
                                unsigned*       __restrict__ Sync) {
    extern __shared__ char smem[];
    _Float16* Wl   = (_Float16*)smem;                          // 64 x 1024 f16 = 131072 B
    float*    gbuf = (float*)(smem + 131072);                  // 64 x 64 f32   = 16384 B
    float*    cst  = (float*)(smem + 131072 + 16384);          // 64 x 16 f32   = 4096 B
    float*    bias = (float*)(smem + 131072 + 16384 + 4096);   // 64 f32

    const int wg  = blockIdx.x;
    const int dir = wg / CGRP;
    const int cg  = wg % CGRP;
    const int tid = threadIdx.x;

    // stage weight slice into LDS (8192 x 16B)
    {
        const v8h* src = (const v8h*)(Wp + (size_t)wg * NROWS * KTOT);
        v8h* dst = (v8h*)Wl;
        for (int i = tid; i < 8192; i += 256) dst[i] = src[i];
    }
    if (tid < NROWS) bias[tid] = Bp[wg * NROWS + tid];
    for (int i = tid; i < BSZ * HCOLS; i += 256) cst[i] = 0.0f;
    __syncthreads();

    const int wid   = tid >> 5;
    const int lane  = tid & 31;
    const int mt    = wid & 3;               // batch tile 0..3
    const int n0    = (wid >> 2) << 1;       // gate-tile pair: {0,1} or {2,3}
    const int arow  = mt * 16 + (lane & 15); // batch row 0..63
    const int ahalf = (lane & 16) ? 8 : 0;   // A fragment K sub-offset
    const int bhalf = (lane & 16) ? 8 : 0;   // B fragment K sub-offset
    const int cmb   = (lane & 16) ? 8 : 0;   // C fragment M base
    const int cn    = lane & 15;             // C fragment N
    const _Float16* b0base = Wl + (size_t)(n0 * 16 + (lane & 15)) * KTOT + bhalf;
    const _Float16* b1base = b0base + (size_t)16 * KTOT;

    for (int t = 0; t < SSQ; ++t) {
        const int bufR = t & 1;
        const _Float16* xrow = Xh  + (((size_t)t * BSZ + arow) << 9) + ahalf;
        const _Float16* hrow = Hst + (((size_t)(bufR * 2 + dir) * BSZ + arow) << 9) + ahalf;

        v8f acc0 = {}; v8f acc1 = {};

        // ---- phase 1: x_t contribution (independent of h_{t-1}; overlaps the
        //      device-wide barrier latency of other workgroups) ----
        #pragma unroll 4
        for (int kt = 0; kt < 16; ++kt) {
            V16U a, b0, b1;
            const _Float16* ap  = xrow + kt * 32;
            a.p.lo  = *(const v8h*)(ap);
            a.p.hi  = *(const v8h*)(ap + 16);
            const _Float16* bp0 = b0base + kt * 32;
            const _Float16* bp1 = b1base + kt * 32;
            b0.p.lo = *(const v8h*)(bp0);
            b0.p.hi = *(const v8h*)(bp0 + 16);
            b1.p.lo = *(const v8h*)(bp1);
            b1.p.hi = *(const v8h*)(bp1 + 16);
            acc0 = __builtin_amdgcn_wmma_f32_16x16x32_f16(false, a.v, false, b0.v, (short)0, acc0, false, false);
            acc1 = __builtin_amdgcn_wmma_f32_16x16x32_f16(false, a.v, false, b1.v, (short)0, acc1, false, false);
        }

        // ---- late wait: h_{t-1} must be published by all WGs before phase 2 ----
        if (t != 0) {
            if (tid == 0) {
                const unsigned target = (unsigned)NWG * (unsigned)t;
                while (__hip_atomic_load(Sync, __ATOMIC_ACQUIRE, __HIP_MEMORY_SCOPE_AGENT) < target) {
                    __builtin_amdgcn_s_sleep(2);
                }
            }
            __syncthreads();
        }

        // ---- phase 2: h_{t-1} contribution ----
        #pragma unroll 4
        for (int kt = 0; kt < 16; ++kt) {
            V16U a, b0, b1;
            const _Float16* ap  = hrow + kt * 32;
            a.p.lo  = *(const v8h*)(ap);
            a.p.hi  = *(const v8h*)(ap + 16);
            const _Float16* bp0 = b0base + (kt + 16) * 32;
            const _Float16* bp1 = b1base + (kt + 16) * 32;
            b0.p.lo = *(const v8h*)(bp0);
            b0.p.hi = *(const v8h*)(bp0 + 16);
            b1.p.lo = *(const v8h*)(bp1);
            b1.p.hi = *(const v8h*)(bp1 + 16);
            acc0 = __builtin_amdgcn_wmma_f32_16x16x32_f16(false, a.v, false, b0.v, (short)0, acc0, false, false);
            acc1 = __builtin_amdgcn_wmma_f32_16x16x32_f16(false, a.v, false, b1.v, (short)0, acc1, false, false);
        }

        // scatter C tiles to gate buffer: gbuf[batch 0..63][gatecol 0..63]
        {
            const int gb = mt * 16 + cmb;
            #pragma unroll
            for (int v = 0; v < 8; ++v) {
                gbuf[(gb + v) * 64 + n0 * 16 + cn]       = acc0[v];
                gbuf[(gb + v) * 64 + (n0 + 1) * 16 + cn] = acc1[v];
            }
        }
        __syncthreads();

        // elementwise LSTM cell for owned (batch, h-col) pairs
        const int bufW = (t + 1) & 1;
        for (int p = tid; p < BSZ * HCOLS; p += 256) {
            int b = p >> 4, j = p & 15;
            float ig = gbuf[b * 64 + j]      + bias[j];
            float fg = gbuf[b * 64 + 16 + j] + bias[16 + j];
            float gg = gbuf[b * 64 + 32 + j] + bias[32 + j];
            float og = gbuf[b * 64 + 48 + j] + bias[48 + j];
            float cnew = fsigmoid(fg) * cst[p] + fsigmoid(ig) * ftanh(gg);
            float h    = fsigmoid(og) * ftanh(cnew);
            cst[p] = cnew;
            int col = cg * HCOLS + j;
            Outs[((size_t)t * BSZ + b) * (2 * HHD) + (size_t)dir * HHD + col] = h;
            size_t o2 = (((size_t)t * 2 + dir) * BSZ + b) * HHD + col;
            Hs[o2] = h;
            Cs[o2] = cnew;
            Hst[(((size_t)(bufW * 2 + dir) * BSZ + b) << 9) + col] = (_Float16)h;
        }

        // ---- early signal: publish h_t, then immediately start next step's
        //      x-phase without waiting for the rest of the device ----
        if (t != SSQ - 1) {
            __threadfence();
            __syncthreads();
            if (tid == 0) {
                __hip_atomic_fetch_add(Sync, 1u, __ATOMIC_RELEASE, __HIP_MEMORY_SCOPE_AGENT);
            }
        }
    }
}

// ---------------- launch ----------------

extern "C" void kernel_launch(void* const* d_in, const int* in_sizes, int n_in,
                              void* d_out, int out_size, void* d_ws, size_t ws_size,
                              hipStream_t stream) {
    const float* X     = (const float*)d_in[0];
    const float* Wih_f = (const float*)d_in[1];
    const float* Whh_f = (const float*)d_in[2];
    const float* bih_f = (const float*)d_in[3];
    const float* bhh_f = (const float*)d_in[4];
    const float* Wih_b = (const float*)d_in[5];
    const float* Whh_b = (const float*)d_in[6];
    const float* bih_b = (const float*)d_in[7];
    const float* bhh_b = (const float*)d_in[8];

    char* ws = (char*)d_ws;
    _Float16* Xh  = (_Float16*)(ws + WS_XH);
    _Float16* Wp  = (_Float16*)(ws + WS_WP);
    float*    Bp  = (float*)   (ws + WS_BP);
    _Float16* Hst = (_Float16*)(ws + WS_HST);
    unsigned* Syn = (unsigned*)(ws + WS_SYNC);

    float* Outs = (float*)d_out;
    float* Hso  = Outs + (size_t)SSQ * BSZ * 2 * HHD;
    float* Cso  = Hso  + (size_t)SSQ * 2 * BSZ * HHD;

    k_init<<<512, 256, 0, stream>>>((unsigned*)Hst, Syn);
    k_convert_x<<<32768, 256, 0, stream>>>(X, Xh);
    k_pack_w<<<4096, 256, 0, stream>>>(Wih_f, Whh_f, Wih_b, Whh_b, Wp);
    k_pack_bias<<<16, 256, 0, stream>>>(bih_f, bhh_f, bih_b, bhh_b, Bp);

    const size_t lds_bytes = 131072 + 16384 + 4096 + 256;
    (void)hipFuncSetAttribute((const void*)lstm_persistent,
                              hipFuncAttributeMaxDynamicSharedMemorySize,
                              (int)lds_bytes);
    lstm_persistent<<<NWG, 256, lds_bytes, stream>>>(Xh, Wp, Bp, Hst, Outs, Hso, Cso, Syn);
}
